// Model_22419729285256
// MI455X (gfx1250) — matrix operand order repaired and checked by reading, not compiled, not run on hardware
//
#include <hip/hip_runtime.h>

typedef __attribute__((ext_vector_type(16))) __bf16 v16bf;
typedef __attribute__((ext_vector_type(8)))  float  v8f;

constexpr int   kFEAT = 923;
constexpr int   kHID  = 64;
constexpr int   kT    = 5;
constexpr int   kB    = 8192;
constexpr float kEPS  = 1e-5f;

// ---- padded bf16 weight region (elements of ushort) ----
constexpr int O_WTD   = 0;                       // td_W   (64 x 928)
constexpr int O_WHIST = O_WTD   + 64  * 928;     // hist_W (928 x 64)
constexpr int O_WENC  = O_WHIST + 928 * 64;      // enc_W  (928 x 64)
constexpr int O_WRNN  = O_WENC  + 928 * 64;      // [rnn_Wih | rnn_Whh] (256 x 1920)
constexpr int O_WATTH = O_WRNN  + 256 * 1920;    // att_W[:, :64]  (64 x 64)
constexpr int O_WATTE = O_WATTH + 64  * 64;      // att_W[:, 64:]  (64 x 928)
constexpr int O_WDEC  = O_WATTE + 64  * 928;     // [dec_Wih | dec_Whh] (256 x 992)
constexpr int NW_TOT  = O_WDEC  + 256 * 992;

// ---- workspace byte offsets ----
constexpr size_t alup(size_t x) { return (x + 255) & ~(size_t)255; }
constexpr size_t O_W     = 0;
constexpr size_t O_GAMMA = alup(O_W     + (size_t)NW_TOT * 2);
constexpr size_t O_H     = alup(O_GAMMA + (size_t)kB * kT * kHID * 4);
constexpr size_t O_C     = alup(O_H     + (size_t)kB * kHID * 4);
constexpr size_t O_HBF   = alup(O_C     + (size_t)kB * kHID * 4);
constexpr size_t O_RNNS  = alup(O_HBF   + (size_t)kB * kHID * 2);
constexpr size_t O_ENCS  = alup(O_RNNS  + (size_t)kB * 1920 * 2);  // also deltas_bf16 early on
constexpr size_t O_E     = alup(O_ENCS  + (size_t)kB * kT * 928 * 2);
constexpr size_t O_HH    = alup(O_E     + (size_t)kB * kT * kHID * 4);
constexpr size_t O_GATES = alup(O_HH    + (size_t)kB * kHID * 4);
constexpr size_t O_DECS  = alup(O_GATES + (size_t)kB * 256 * 4);
constexpr size_t O_YH    = alup(O_DECS  + (size_t)kB * 992 * 2);
constexpr size_t O_ATTN  = alup(O_YH    + (size_t)kB * 2 * 4);
constexpr size_t O_ACC   = alup(O_ATTN  + (size_t)kB * kT * 4);

__device__ __forceinline__ unsigned short f2bf(float f) {
  return __builtin_bit_cast(unsigned short, (__bf16)f);   // native v_cvt, RNE
}
__device__ __forceinline__ float bf2f(unsigned short h) {
  return (float)__builtin_bit_cast(__bf16, h);
}
__device__ __forceinline__ float sigm(float x)   { return 1.0f / (1.0f + expf(-x)); }
__device__ __forceinline__ float round6(float a) { return rintf(a * 1.0e6f) / 1.0e6f; }

// ---------------- weight build: fp32 -> padded bf16 ----------------
__global__ void build_weights(const float* td_W, const float* hist_W, const float* enc_W,
                              const float* rnn_Wih, const float* rnn_Whh,
                              const float* att_W, const float* dec_Wih, const float* dec_Whh,
                              unsigned short* Wb) {
  int i = blockIdx.x * blockDim.x + threadIdx.x;
  if (i >= NW_TOT) return;
  float v = 0.f;
  if (i < O_WHIST)      { int r = i / 928, c = i % 928; if (c < kFEAT) v = td_W[r * kFEAT + c]; }
  else if (i < O_WENC)  { int j = i - O_WHIST; int r = j / 64, c = j % 64; if (r < kFEAT) v = hist_W[r * 64 + c]; }
  else if (i < O_WRNN)  { int j = i - O_WENC;  int r = j / 64, c = j % 64; if (r < kFEAT) v = enc_W[r * 64 + c]; }
  else if (i < O_WATTH) { int j = i - O_WRNN;  int r = j / 1920, c = j % 1920;
                          if (c < 1846) v = rnn_Wih[r * 1846 + c];
                          else if (c < 1910) v = rnn_Whh[r * 64 + (c - 1846)]; }
  else if (i < O_WATTE) { int j = i - O_WATTH; int r = j / 64, c = j % 64; v = att_W[r * 987 + c]; }
  else if (i < O_WDEC)  { int j = i - O_WATTE; int r = j / 928, c = j % 928; if (c < kFEAT) v = att_W[r * 987 + 64 + c]; }
  else                  { int j = i - O_WDEC;  int r = j / 992, c = j % 992;
                          if (c < 925) v = dec_Wih[r * 925 + c];
                          else if (c < 989) v = dec_Whh[r * 64 + (c - 925)]; }
  Wb[i] = f2bf(v);
}

__global__ void init_state(float* h, float* c, float* acc) {
  int i = blockIdx.x * blockDim.x + threadIdx.x;
  if (i < kB * kHID) { h[i] = 0.f; c[i] = 0.f; }
  if (i < 32) acc[i] = 0.f;
}

// deltas (B*T x 923 f32) -> padded bf16 (B*T x 928), zero pad cols
__global__ void cvt_deltas(const float* deltas, unsigned short* dst) {
  size_t i = (size_t)blockIdx.x * blockDim.x + threadIdx.x;
  if (i >= (size_t)kB * kT * 928) return;
  size_t r = i / 928; int c = (int)(i % 928);
  dst[i] = f2bf(c < kFEAT ? deltas[r * kFEAT + c] : 0.f);
}

// ---------------- generic WMMA GEMM: C = A (MxK,bf16) * W^T (NxK,bf16), fused epilogues ----
enum { EPI_GAMMA, EPI_XH, EPI_ENC, EPI_BIAS, EPI_NONE };

struct GemmP {
  const unsigned short* A; int lda;  // bf16 row-major, lda in elements, K padded to 32
  const unsigned short* W; int ldw;  // bf16 row-major (N rows x K cols), K padded
  int M, N, K;                       // K multiple of 32, M multiple of 128
  const float* bias;
  float* outf; int ldo;
  const float* values;
  const float* masks;
  float* imps;
  unsigned short* stage; int stage_ld;
  int t;
  float* ss; float* sm;
};

template<int EPI>
__global__ __launch_bounds__(128)
void gemm_wmma(GemmP p) {
  const int lane = threadIdx.x & 31;
  const int wave = threadIdx.x >> 5;
  const int m0 = (blockIdx.y * 4 + wave) * 32;   // 32-row wave tile
  const int n0 = blockIdx.x * 32;                // 32-col wave tile
  if (m0 >= p.M) return;
  const int lg = lane >> 4;      // lane group 0/1
  const int li = lane & 15;

  union AU { v16bf v; unsigned int u[8]; };
  AU a0, a1, b0, b1;
  v8f acc00 = {0.f,0.f,0.f,0.f,0.f,0.f,0.f,0.f};
  v8f acc01 = acc00, acc10 = acc00, acc11 = acc00;

  const unsigned short* Ar0 = p.A + (size_t)(m0 + li)      * p.lda;
  const unsigned short* Ar1 = p.A + (size_t)(m0 + 16 + li) * p.lda;
  const unsigned short* Wr0 = p.W + (size_t)(n0 + li)      * p.ldw;
  const unsigned short* Wr1 = p.W + (size_t)(n0 + 16 + li) * p.ldw;
  const int kchunks = p.K >> 5;

  for (int kc = 0; kc < kchunks; ++kc) {
    const int kbA = kc * 32 + (lg << 4);   // A: lanes 0-15 -> K+0..15 ; lanes 16-31 -> K+16..31
    const int kbB = kc * 32 + (lg << 4);   // B: lanes 0-15 -> K+0..15 ; lanes 16-31 -> K+16..31
    {
      const unsigned int* q0 = (const unsigned int*)(Ar0 + kbA);
      const unsigned int* q1 = (const unsigned int*)(Ar0 + kbA + 8);
      const unsigned int* q2 = (const unsigned int*)(Ar1 + kbA);
      const unsigned int* q3 = (const unsigned int*)(Ar1 + kbA + 8);
#pragma unroll
      for (int i = 0; i < 4; ++i) {
        a0.u[i] = q0[i]; a0.u[4 + i] = q1[i];
        a1.u[i] = q2[i]; a1.u[4 + i] = q3[i];
      }
    }
    {
      const unsigned int* q0 = (const unsigned int*)(Wr0 + kbB);
      const unsigned int* q1 = (const unsigned int*)(Wr1 + kbB);
#pragma unroll
      for (int i = 0; i < 8; ++i) { b0.u[i] = q0[i]; b1.u[i] = q1[i]; }
    }
    acc00 = __builtin_amdgcn_wmma_f32_16x16x32_bf16(false, a0.v, false, b0.v, (short)0, acc00, false, false);
    acc01 = __builtin_amdgcn_wmma_f32_16x16x32_bf16(false, a0.v, false, b1.v, (short)0, acc01, false, false);
    acc10 = __builtin_amdgcn_wmma_f32_16x16x32_bf16(false, a1.v, false, b0.v, (short)0, acc10, false, false);
    acc11 = __builtin_amdgcn_wmma_f32_16x16x32_bf16(false, a1.v, false, b1.v, (short)0, acc11, false, false);
  }

  float lss = 0.f, lsm = 0.f;
#pragma unroll
  for (int ms = 0; ms < 2; ++ms) {
#pragma unroll
    for (int s = 0; s < 2; ++s) {
      const int nn = n0 + s * 16 + li;
      if (nn < p.N) {
        v8f av = ms ? (s ? acc11 : acc10) : (s ? acc01 : acc00);
#pragma unroll
        for (int r = 0; r < 8; ++r) {
          const int mm = m0 + ms * 16 + r + (lg << 3);
          float v = av[r];
          if constexpr (EPI == EPI_GAMMA) {
            p.outf[(size_t)mm * p.ldo + nn] = expf(-fmaxf(v + p.bias[nn], 0.f));
          } else if constexpr (EPI == EPI_XH) {
            float xh = v + p.bias[nn];
            size_t vi = ((size_t)mm * kT + p.t) * kFEAT + nn;
            float x = p.values[vi], m = p.masks[vi];
            float d = (x - xh) * m;
            lss += d * d; lsm += m;
            float xc = m * x + (1.f - m) * xh;
            p.imps[vi] = round6(xc);
            unsigned short* st = p.stage + (size_t)mm * p.stage_ld;
            st[nn]         = f2bf(xc);
            st[kFEAT + nn] = f2bf(m);
          } else if constexpr (EPI == EPI_ENC) {
            size_t vi = ((size_t)mm * kT + p.t) * kFEAT + nn;
            float e = tanhf(v + p.bias[nn]) * p.masks[vi];
            p.stage[((size_t)mm * kT + p.t) * p.stage_ld + nn] = f2bf(e);
          } else if constexpr (EPI == EPI_BIAS) {
            p.outf[(size_t)mm * p.ldo + nn] = v + p.bias[nn];
          } else {
            p.outf[(size_t)mm * p.ldo + nn] = v;
          }
        }
      }
    }
  }
  if constexpr (EPI == EPI_XH) {
#pragma unroll
    for (int off = 16; off > 0; off >>= 1) {
      lss += __shfl_down(lss, off, 32);
      lsm += __shfl_down(lsm, off, 32);
    }
    if (lane == 0) { atomicAdd(p.ss, lss); atomicAdd(p.sm, lsm); }
  }
}

// ---------------- elementwise kernels ----------------
__global__ void gammamul(float* h, const float* gamma, unsigned short* hbf,
                         unsigned short* rnnS, unsigned short* encS, int t) {
  int i = blockIdx.x * blockDim.x + threadIdx.x;
  if (i >= kB * kHID) return;
  int b = i >> 6, j = i & 63;
  float hv = h[i] * gamma[((size_t)b * kT + t) * kHID + j];
  h[i] = hv;
  unsigned short hb = f2bf(hv);
  hbf[i] = hb;
  rnnS[(size_t)b * 1920 + 1846 + j] = hb;
  if (j < 10) rnnS[(size_t)b * 1920 + 1910 + j] = 0;                 // K pad
  if (j < 5)  encS[((size_t)b * kT + t) * 928 + 923 + j] = 0;        // K pad
}

__global__ void lstm_k(const float* gates, float* h, float* c, unsigned short* hbf) {
  int i = blockIdx.x * blockDim.x + threadIdx.x;
  if (i >= kB * kHID) return;
  int b = i >> 6, j = i & 63;
  const float* g = gates + (size_t)b * 256;
  float ii = sigm(g[j]), ff = sigm(g[64 + j]), gg = tanhf(g[128 + j]), oo = sigm(g[192 + j]);
  float c2 = ff * c[i] + ii * gg;
  float h2 = oo * tanhf(c2);
  c[i] = c2; h[i] = h2; hbf[i] = f2bf(h2);
}

__global__ void y_head(const float* h, const float* out_W, const float* out_b,
                       const float* labels, const float* masks_y,
                       float* yh, float* decd, float* ss, float* sm, int td, int write_dec) {
  int b = blockIdx.x * blockDim.x + threadIdx.x;
  float lss = 0.f, lsm = 0.f;
  if (b < kB) {
    const float* hb = h + (size_t)b * kHID;
    float a0 = out_b[0], a1 = out_b[1];
    for (int j = 0; j < kHID; ++j) { float hv = hb[j]; a0 += hv * out_W[j]; a1 += hv * out_W[kHID + j]; }
    yh[b * 2] = a0; yh[b * 2 + 1] = a1;
    int tsrc = kT - 1 - td;
    for (int k = 0; k < 2; ++k) {
      float y = k ? a1 : a0;
      float l = labels [((size_t)b * kT + tsrc) * 2 + k];
      float m = masks_y[((size_t)b * kT + tsrc) * 2 + k];
      float d = (y - l) * m;
      lss += d * d; lsm += m;
      if (write_dec) decd[((size_t)b * kT + tsrc) * 2 + k] = round6(l * m + y * (1.f - m));
    }
  }
#pragma unroll
  for (int off = 16; off > 0; off >>= 1) {
    lss += __shfl_down(lss, off, 32);
    lsm += __shfl_down(lsm, off, 32);
  }
  if ((threadIdx.x & 31) == 0) { atomicAdd(ss + td, lss); atomicAdd(sm + td, lsm); }
}

__global__ void attn_k(const float* Hh, const float* E, const float* att_v, float* attn) {
  int b = blockIdx.x * blockDim.x + threadIdx.x;
  if (b >= kB) return;
  const float* hh = Hh + (size_t)b * kHID;
  float s[kT], mx = -1e30f;
  for (int t = 0; t < kT; ++t) {
    const float* e = E + ((size_t)b * kT + t) * kHID;
    float a = 0.f;
    for (int j = 0; j < kHID; ++j) a += att_v[j] * tanhf(hh[j] + e[j]);
    s[t] = a; mx = fmaxf(mx, a);
  }
  float den = 0.f;
  for (int t = 0; t < kT; ++t) { s[t] = expf(s[t] - mx); den += s[t]; }
  for (int t = 0; t < kT; ++t) attn[(size_t)b * kT + t] = s[t] / den;
}

__global__ void input_y_k(const float* yh, const float* labels, const float* masks_y,
                          unsigned short* decS, float* decd, int td) {
  int b = blockIdx.x * blockDim.x + threadIdx.x;
  if (b >= kB) return;
  int tsrc = kT - 1 - td;
  for (int k = 0; k < 2; ++k) {
    float l = labels [((size_t)b * kT + tsrc) * 2 + k];
    float m = masks_y[((size_t)b * kT + tsrc) * 2 + k];
    float iy = l * m + yh[b * 2 + k] * (1.f - m);
    decS[(size_t)b * 992 + k] = f2bf(iy);
    decd[((size_t)b * kT + tsrc) * 2 + k] = round6(iy);
  }
}

__global__ void context_k(const float* attn, const unsigned short* encS,
                          const unsigned short* hbf, unsigned short* decS) {
  int i = blockIdx.x * blockDim.x + threadIdx.x;
  int b = i / 990, d = i % 990;
  if (b >= kB) return;
  if (d < kFEAT) {
    float s = 0.f;
    for (int t = 0; t < kT; ++t)
      s += attn[(size_t)b * kT + t] * bf2f(encS[((size_t)b * kT + t) * 928 + d]);
    decS[(size_t)b * 992 + 2 + d] = f2bf(s);
  } else if (d < kFEAT + kHID) {
    int j = d - kFEAT;
    decS[(size_t)b * 992 + 925 + j] = hbf[(size_t)b * kHID + j];
  } else {
    decS[(size_t)b * 992 + 989 + (d - (kFEAT + kHID))] = 0;   // K pad
  }
}

__global__ void finalize_k(const float* acc, float* out) {
  if (threadIdx.x != 0 || blockIdx.x != 0) return;
  float xl = 0.f, yl = 0.f;
  for (int t = 0; t < kT; ++t) xl += (acc[t]      / (float)(kB * kFEAT)) / (acc[5 + t]  + kEPS);
  for (int t = 0; t < kT; ++t) yl += (acc[10 + t] / (float)(kB * 2))     / (acc[15 + t] + kEPS);
  out[0] = xl / kT + yl / kT;
}

// ---------------- host launch ----------------
extern "C" void kernel_launch(void* const* d_in, const int* in_sizes, int n_in,
                              void* d_out, int out_size, void* d_ws, size_t ws_size,
                              hipStream_t stream) {
  (void)in_sizes; (void)n_in; (void)out_size; (void)ws_size;
  const float* values  = (const float*)d_in[0];
  const float* masks   = (const float*)d_in[1];
  const float* deltas  = (const float*)d_in[2];
  const float* labels  = (const float*)d_in[3];
  const float* masks_y = (const float*)d_in[4];
  const float* td_W    = (const float*)d_in[5];
  const float* td_b    = (const float*)d_in[6];
  const float* hist_W  = (const float*)d_in[7];
  const float* hist_b  = (const float*)d_in[8];
  const float* enc_W   = (const float*)d_in[9];
  const float* enc_b   = (const float*)d_in[10];
  const float* rnn_Wih = (const float*)d_in[11];
  const float* rnn_Whh = (const float*)d_in[12];
  const float* rnn_b   = (const float*)d_in[13];
  const float* dec_Wih = (const float*)d_in[14];
  const float* dec_Whh = (const float*)d_in[15];
  const float* dec_b   = (const float*)d_in[16];
  const float* att_W   = (const float*)d_in[17];
  const float* att_b   = (const float*)d_in[18];
  const float* att_v   = (const float*)d_in[19];
  const float* out_W   = (const float*)d_in[20];
  const float* out_b   = (const float*)d_in[21];

  char* ws = (char*)d_ws;
  unsigned short* Wb  = (unsigned short*)(ws + O_W);
  float* gamma        = (float*)(ws + O_GAMMA);
  float* hbuf         = (float*)(ws + O_H);
  float* cbuf         = (float*)(ws + O_C);
  unsigned short* hbf = (unsigned short*)(ws + O_HBF);
  unsigned short* rnnS= (unsigned short*)(ws + O_RNNS);
  unsigned short* encS= (unsigned short*)(ws + O_ENCS);  // deltas_bf16 first, enc later
  float* Ebuf         = (float*)(ws + O_E);
  float* Hh           = (float*)(ws + O_HH);
  float* gates        = (float*)(ws + O_GATES);
  unsigned short* decS= (unsigned short*)(ws + O_DECS);
  float* yh           = (float*)(ws + O_YH);
  float* attn         = (float*)(ws + O_ATTN);
  float* acc          = (float*)(ws + O_ACC);

  float* outF = (float*)d_out;
  float* imps = outF + 1;
  float* decd = outF + 1 + (size_t)kB * kT * kFEAT;

  init_state<<<(kB * kHID + 255) / 256, 256, 0, stream>>>(hbuf, cbuf, acc);
  build_weights<<<(NW_TOT + 255) / 256, 256, 0, stream>>>(td_W, hist_W, enc_W, rnn_Wih, rnn_Whh,
                                                          att_W, dec_Wih, dec_Whh, Wb);
  cvt_deltas<<<((kB * kT * 928) + 255) / 256, 256, 0, stream>>>(deltas, encS);
  {  // gamma for all (b,t): exp(-relu(deltas @ td_W.T + td_b)), bf16 WMMA path
    GemmP p{}; p.A = encS; p.lda = 928; p.W = Wb + O_WTD; p.ldw = 928;
    p.M = kB * kT; p.N = kHID; p.K = 928;
    p.bias = td_b; p.outf = gamma; p.ldo = kHID;
    gemm_wmma<EPI_GAMMA><<<dim3(2, (kB * kT) / 128), 128, 0, stream>>>(p);
  }
  for (int t = 0; t < kT; ++t) {
    gammamul<<<(kB * kHID + 255) / 256, 256, 0, stream>>>(hbuf, gamma, hbf, rnnS, encS, t);
    {  // x_h GEMM with fused x_c / imputation / x-loss / RNN staging
      GemmP p{}; p.A = hbf; p.lda = kHID; p.W = Wb + O_WHIST; p.ldw = kHID;
      p.M = kB; p.N = kFEAT; p.K = 64;
      p.bias = hist_b; p.values = values; p.masks = masks; p.imps = imps;
      p.stage = rnnS; p.stage_ld = 1920; p.t = t; p.ss = acc + t; p.sm = acc + 5 + t;
      gemm_wmma<EPI_XH><<<dim3(29, kB / 128), 128, 0, stream>>>(p);
    }
    {  // enc = tanh(h@enc_W.T + b) * m, stored bf16 for attention & context
      GemmP p{}; p.A = hbf; p.lda = kHID; p.W = Wb + O_WENC; p.ldw = kHID;
      p.M = kB; p.N = kFEAT; p.K = 64;
      p.bias = enc_b; p.masks = masks; p.stage = encS; p.stage_ld = 928; p.t = t;
      gemm_wmma<EPI_ENC><<<dim3(29, kB / 128), 128, 0, stream>>>(p);
    }
    {  // fused gates = [x_c, m, h] @ [Wih | Whh].T + b
      GemmP p{}; p.A = rnnS; p.lda = 1920; p.W = Wb + O_WRNN; p.ldw = 1920;
      p.M = kB; p.N = 256; p.K = 1920;
      p.bias = rnn_b; p.outf = gates; p.ldo = 256;
      gemm_wmma<EPI_BIAS><<<dim3(8, kB / 128), 128, 0, stream>>>(p);
    }
    lstm_k<<<(kB * kHID + 255) / 256, 256, 0, stream>>>(gates, hbuf, cbuf, hbf);
  }
  {  // attention precompute: E = enc_out @ att_We.T + att_b (constant over decoder steps)
    GemmP p{}; p.A = encS; p.lda = 928; p.W = Wb + O_WATTE; p.ldw = 928;
    p.M = kB * kT; p.N = kHID; p.K = 928;
    p.bias = att_b; p.outf = Ebuf; p.ldo = kHID;
    gemm_wmma<EPI_BIAS><<<dim3(2, (kB * kT) / 128), 128, 0, stream>>>(p);
  }
  y_head<<<(kB + 255) / 256, 256, 0, stream>>>(hbuf, out_W, out_b, labels, masks_y,
                                               yh, decd, acc + 10, acc + 15, 0, 1);
  for (int td = 1; td < kT; ++td) {
    {  // Hh = h @ att_Wh.T
      GemmP p{}; p.A = hbf; p.lda = kHID; p.W = Wb + O_WATTH; p.ldw = kHID;
      p.M = kB; p.N = kHID; p.K = 64;
      p.outf = Hh; p.ldo = kHID;
      gemm_wmma<EPI_NONE><<<dim3(2, kB / 128), 128, 0, stream>>>(p);
    }
    attn_k<<<(kB + 255) / 256, 256, 0, stream>>>(Hh, Ebuf, att_v, attn);
    input_y_k<<<(kB + 255) / 256, 256, 0, stream>>>(yh, labels, masks_y, decS, decd, td);
    context_k<<<(kB * 990 + 255) / 256, 256, 0, stream>>>(attn, encS, hbf, decS);
    {  // fused gates = [input_y, context, h] @ [Wih | Whh].T + b
      GemmP p{}; p.A = decS; p.lda = 992; p.W = Wb + O_WDEC; p.ldw = 992;
      p.M = kB; p.N = 256; p.K = 992;
      p.bias = dec_b; p.outf = gates; p.ldo = 256;
      gemm_wmma<EPI_BIAS><<<dim3(8, kB / 128), 128, 0, stream>>>(p);
    }
    lstm_k<<<(kB * kHID + 255) / 256, 256, 0, stream>>>(gates, hbuf, cbuf, hbf);
    y_head<<<(kB + 255) / 256, 256, 0, stream>>>(hbuf, out_W, out_b, labels, masks_y,
                                                 yh, decd, acc + 10, acc + 15, td, 0);
  }
  finalize_k<<<1, 1, 0, stream>>>(acc, outF);
}
